// CPT_20529943675022
// MI455X (gfx1250) — matrix operand order repaired and checked by reading, not compiled, not run on hardware
//
#include <hip/hip_runtime.h>
#include <stdint.h>

// ---------- types ----------
typedef __attribute__((ext_vector_type(16))) __bf16   v16bf;
typedef __attribute__((ext_vector_type(8)))  float    v8f;
typedef __attribute__((ext_vector_type(8)))  uint32_t v8u;

// ---------- helpers ----------
__device__ __forceinline__ uint16_t f2bf(float f) {
    uint32_t u = __builtin_bit_cast(uint32_t, f);
    u += 0x7FFFu + ((u >> 16) & 1u);          // round-to-nearest-even
    return (uint16_t)(u >> 16);
}
__device__ __forceinline__ uint32_t pack2(float lo, float hi) {
    return (uint32_t)f2bf(lo) | ((uint32_t)f2bf(hi) << 16);
}
__device__ __forceinline__ v8f wmma_bf16(v8u a, v8u b, v8f c) {
    return __builtin_amdgcn_wmma_f32_16x16x32_bf16(
        false, __builtin_bit_cast(v16bf, a),
        false, __builtin_bit_cast(v16bf, b),
        (short)0, c, false, false);
}
// A fragment (16x32 bf16): base = row_ptr_u32 + k0/2 + (lane/16)*8
__device__ __forceinline__ v8u load_a(const uint32_t* base) {
    v8u r;
    r.s0 = base[0];  r.s1 = base[1];  r.s2 = base[2];  r.s3 = base[3];
    r.s4 = base[4];  r.s5 = base[5];  r.s6 = base[6];  r.s7 = base[7];
    return r;
}
// B fragment (32x16 bf16): 8 contiguous u32
__device__ __forceinline__ v8u load_b(const uint32_t* base) {
    v8u r;
    r.s0 = base[0]; r.s1 = base[1]; r.s2 = base[2]; r.s3 = base[3];
    r.s4 = base[4]; r.s5 = base[5]; r.s6 = base[6]; r.s7 = base[7];
    return r;
}
// butterfly ops via ds_swizzle (group-of-32, and=0x1F, xor in offset[14:10])
#define SWZ_F32(x, imm) \
    __builtin_bit_cast(float, __builtin_amdgcn_ds_swizzle(__builtin_bit_cast(int, (x)), (imm)))
__device__ __forceinline__ float half16_max(float v) {
    v = fmaxf(v, SWZ_F32(v, 0x041F));
    v = fmaxf(v, SWZ_F32(v, 0x081F));
    v = fmaxf(v, SWZ_F32(v, 0x101F));
    v = fmaxf(v, SWZ_F32(v, 0x201F));
    return v;
}
__device__ __forceinline__ float half16_sum(float v) {
    v += SWZ_F32(v, 0x041F);
    v += SWZ_F32(v, 0x081F);
    v += SWZ_F32(v, 0x101F);
    v += SWZ_F32(v, 0x201F);
    return v;
}
__device__ __forceinline__ float fast_tanh(float x) {
    float e = __expf(2.0f * x);
    return 1.0f - 2.0f * __builtin_amdgcn_rcpf(e + 1.0f);
}

// ---------- problem constants ----------
#define NB 32
#define NS 1024
#define NT 64
#define ND 1024    // 2H
#define NF 2048    // 4H
#define NROWS (NB * NS)            // 32768
#define CAT_U32 (NF / 2)           // 1024 u32 per cat row
#define D_U32  (ND / 2)            // 512

// ---------- prep kernels (f32 -> bf16 layouts) ----------
__global__ void prep_hidden_kernel(const float* __restrict__ h, uint32_t* __restrict__ catU) {
    const int n = NROWS * D_U32;
    for (int i = blockIdx.x * blockDim.x + threadIdx.x; i < n; i += gridDim.x * blockDim.x) {
        int row = i >> 9, dp = i & 511;
        const float* s = h + (size_t)row * ND + 2 * dp;
        catU[(size_t)row * CAT_U32 + dp] = pack2(s[0], s[1]);
    }
}
__global__ void prep_tgt_kernel(const float* __restrict__ t, uint32_t* __restrict__ dst) {
    const int n = NB * NT * D_U32;
    for (int i = blockIdx.x * blockDim.x + threadIdx.x; i < n; i += gridDim.x * blockDim.x)
        dst[i] = pack2(t[2 * i], t[2 * i + 1]);
}
__global__ void prep_tgtT_kernel(const float* __restrict__ t, uint32_t* __restrict__ dst) {
    const int n = NB * ND * (NT / 2);
    for (int i = blockIdx.x * blockDim.x + threadIdx.x; i < n; i += gridDim.x * blockDim.x) {
        int b = i >> 15, rem = i & 32767;
        int d = rem >> 5, tp = rem & 31;
        const float* s = t + ((size_t)b * NT + 2 * tp) * ND + d;
        dst[i] = pack2(s[0], s[ND]);
    }
}
__global__ void prep_wT_kernel(const float* __restrict__ w, uint32_t* __restrict__ dst) {
    const int n = ND * (NF / 2);
    for (int i = blockIdx.x * blockDim.x + threadIdx.x; i < n; i += gridDim.x * blockDim.x) {
        int d = i >> 10, fp = i & 1023;
        dst[i] = pack2(w[(size_t)(2 * fp) * ND + d], w[(size_t)(2 * fp + 1) * ND + d]);
    }
}

// ---------- fused attention: scores -> softmax -> ti, writes cat[:,1024:2048] ----------
__global__ __launch_bounds__(128) void attn_kernel(
    uint32_t* __restrict__ catU,          // (NROWS x 1024 u32); reads half0, writes half1
    const uint32_t* __restrict__ tgt,     // (B*T x 512 u32)  row-major T x D bf16
    const uint32_t* __restrict__ tgtT)    // (B*D x 32 u32)   row-major D x T bf16
{
    __shared__ float lds[4][16][68];
    const int lane = threadIdx.x & 31;
    const int wave = threadIdx.x >> 5;
    const int waveId = blockIdx.x * 4 + wave;
    const int b  = waveId >> 6;
    const int s0 = (waveId & 63) << 4;
    const int half = lane >> 4;
    const int ln   = lane & 15;

    // ---- scores: 16 (s) x 64 (t), K over D=1024, ping-pong pipelined ----
    v8f acc[4] = {v8f{}, v8f{}, v8f{}, v8f{}};
    const uint32_t* aRow = catU + (size_t)(b * NS + s0 + ln) * CAT_U32 + half * 8;
    const uint32_t* tCol[4];
#pragma unroll
    for (int j = 0; j < 4; ++j)
        tCol[j] = tgt + (size_t)b * NT * D_U32 + (size_t)(j * 16 + ln) * D_U32 + half * 8;

    v8u a0 = load_a(aRow);
    v8u b0[4];
#pragma unroll
    for (int j = 0; j < 4; ++j) b0[j] = load_b(tCol[j]);

    int k0 = 0;
    for (; k0 + 64 < ND; k0 += 64) {
        const int k1 = (k0 + 32) >> 1, k2 = (k0 + 64) >> 1;
        v8u a1 = load_a(aRow + k1);
        v8u b1[4];
#pragma unroll
        for (int j = 0; j < 4; ++j) b1[j] = load_b(tCol[j] + k1);
#pragma unroll
        for (int j = 0; j < 4; ++j) acc[j] = wmma_bf16(a0, b0[j], acc[j]);
        a0 = load_a(aRow + k2);
#pragma unroll
        for (int j = 0; j < 4; ++j) b0[j] = load_b(tCol[j] + k2);
#pragma unroll
        for (int j = 0; j < 4; ++j) acc[j] = wmma_bf16(a1, b1[j], acc[j]);
    }
    {   // tail: steps k0 and k0+32
        const int k1 = (k0 + 32) >> 1;
        v8u a1 = load_a(aRow + k1);
        v8u b1[4];
#pragma unroll
        for (int j = 0; j < 4; ++j) b1[j] = load_b(tCol[j] + k1);
#pragma unroll
        for (int j = 0; j < 4; ++j) acc[j] = wmma_bf16(a0, b0[j], acc[j]);
#pragma unroll
        for (int j = 0; j < 4; ++j) acc[j] = wmma_bf16(a1, b1[j], acc[j]);
    }

    // ---- row softmax over 64 cols (4 tiles + 16-lane half-wave, ds_swizzle) ----
#pragma unroll
    for (int r = 0; r < 8; ++r) {
        float m = fmaxf(fmaxf(acc[0][r], acc[1][r]), fmaxf(acc[2][r], acc[3][r]));
        m = half16_max(m);
        float s = 0.f;
#pragma unroll
        for (int j = 0; j < 4; ++j) { float e = __expf(acc[j][r] - m); acc[j][r] = e; s += e; }
        s = half16_sum(s);
        float inv = __builtin_amdgcn_rcpf(s);
#pragma unroll
        for (int j = 0; j < 4; ++j) acc[j][r] *= inv;
    }

    // ---- re-layout attn (C layout -> A layout) through LDS ----
#pragma unroll
    for (int j = 0; j < 4; ++j)
#pragma unroll
        for (int r = 0; r < 8; ++r)
            lds[wave][r + 8 * half][j * 16 + ln] = acc[j][r];

    v8u aAt0, aAt1;
#pragma unroll
    for (int p = 0; p < 8; ++p) {
        int kk = half * 16 + 2 * p;
        aAt0[p] = pack2(lds[wave][ln][kk],      lds[wave][ln][kk + 1]);
        aAt1[p] = pack2(lds[wave][ln][32 + kk], lds[wave][ln][32 + kk + 1]);
    }

    // ---- ti = attn @ target : 16 (s) x 1024 (d), K = 64, ping-pong over d tiles ----
    const uint32_t* tT = tgtT + (size_t)b * ND * (NT / 2);
    uint16_t* catH = (uint16_t*)catU;
    const int rowBase = b * NS + s0;

    v8u t00 = load_b(tT + (size_t)ln * 32 + half * 8);
    v8u t01 = load_b(tT + (size_t)ln * 32 + 16 + half * 8);
    int d0 = 0;
    for (; d0 + 32 < ND; d0 += 32) {
        const uint32_t* nb1 = tT + (size_t)(d0 + 16 + ln) * 32 + half * 8;
        v8u t10 = load_b(nb1);
        v8u t11 = load_b(nb1 + 16);
        v8f c = v8f{};
        c = wmma_bf16(aAt0, t00, c);
        c = wmma_bf16(aAt1, t01, c);
#pragma unroll
        for (int r = 0; r < 8; ++r)
            catH[(size_t)(rowBase + r + 8 * half) * NF + ND + d0 + ln] = f2bf(c[r]);

        const uint32_t* nb0 = tT + (size_t)(d0 + 32 + ln) * 32 + half * 8;
        t00 = load_b(nb0);
        t01 = load_b(nb0 + 16);
        v8f c2 = v8f{};
        c2 = wmma_bf16(aAt0, t10, c2);
        c2 = wmma_bf16(aAt1, t11, c2);
#pragma unroll
        for (int r = 0; r < 8; ++r)
            catH[(size_t)(rowBase + r + 8 * half) * NF + ND + d0 + 16 + ln] = f2bf(c2[r]);
    }
    {   // tail: tiles d0 and d0+16
        const uint32_t* nb1 = tT + (size_t)(d0 + 16 + ln) * 32 + half * 8;
        v8u t10 = load_b(nb1);
        v8u t11 = load_b(nb1 + 16);
        v8f c = v8f{};
        c = wmma_bf16(aAt0, t00, c);
        c = wmma_bf16(aAt1, t01, c);
#pragma unroll
        for (int r = 0; r < 8; ++r)
            catH[(size_t)(rowBase + r + 8 * half) * NF + ND + d0 + ln] = f2bf(c[r]);
        v8f c2 = v8f{};
        c2 = wmma_bf16(aAt0, t10, c2);
        c2 = wmma_bf16(aAt1, t11, c2);
#pragma unroll
        for (int r = 0; r < 8; ++r)
            catH[(size_t)(rowBase + r + 8 * half) * NF + ND + d0 + 16 + ln] = f2bf(c2[r]);
    }
}

// ---------- big GEMM: out = tanh(cat @ W + b) + hidden, ping-pong K loop ----------
__global__ __launch_bounds__(128) void gemm_kernel(
    const uint32_t* __restrict__ catU,   // NROWS x 1024 u32 (2048 bf16)
    const uint32_t* __restrict__ wT,     // ND x 1024 u32 (2048 bf16)  == W transposed
    const float* __restrict__ bias,
    const float* __restrict__ hidden,
    float* __restrict__ out)
{
    const int lane = threadIdx.x & 31;
    const int wave = threadIdx.x >> 5;
    const int half = lane >> 4;
    const int ln   = lane & 15;
    const int r0 = blockIdx.x * 64 + wave * 16;
    const int c0 = blockIdx.y * 64;

    v8f acc[4] = {v8f{}, v8f{}, v8f{}, v8f{}};
    const uint32_t* aRow = catU + (size_t)(r0 + ln) * CAT_U32 + half * 8;
    const uint32_t* wCol[4];
#pragma unroll
    for (int j = 0; j < 4; ++j)
        wCol[j] = wT + (size_t)(c0 + j * 16 + ln) * CAT_U32 + half * 8;

    v8u a0 = load_a(aRow);
    v8u b0[4];
#pragma unroll
    for (int j = 0; j < 4; ++j) b0[j] = load_b(wCol[j]);

    int k0 = 0;
    for (; k0 + 64 < NF; k0 += 64) {
        const int k1 = (k0 + 32) >> 1, k2 = (k0 + 64) >> 1;
        __builtin_prefetch((const void*)(aRow + k2 + 64), 0, 0);
        v8u a1 = load_a(aRow + k1);
        v8u b1[4];
#pragma unroll
        for (int j = 0; j < 4; ++j) b1[j] = load_b(wCol[j] + k1);
#pragma unroll
        for (int j = 0; j < 4; ++j) acc[j] = wmma_bf16(a0, b0[j], acc[j]);
        a0 = load_a(aRow + k2);
#pragma unroll
        for (int j = 0; j < 4; ++j) b0[j] = load_b(wCol[j] + k2);
#pragma unroll
        for (int j = 0; j < 4; ++j) acc[j] = wmma_bf16(a1, b1[j], acc[j]);
    }
    {   // tail: steps k0 and k0+32
        const int k1 = (k0 + 32) >> 1;
        v8u a1 = load_a(aRow + k1);
        v8u b1[4];
#pragma unroll
        for (int j = 0; j < 4; ++j) b1[j] = load_b(wCol[j] + k1);
#pragma unroll
        for (int j = 0; j < 4; ++j) acc[j] = wmma_bf16(a0, b0[j], acc[j]);
#pragma unroll
        for (int j = 0; j < 4; ++j) acc[j] = wmma_bf16(a1, b1[j], acc[j]);
    }

#pragma unroll
    for (int j = 0; j < 4; ++j) {
        const int c = c0 + j * 16 + ln;
        const float bbias = bias[c];
#pragma unroll
        for (int r = 0; r < 8; ++r) {
            const int row = r0 + r + 8 * half;
            const size_t idx = (size_t)row * ND + c;
            out[idx] = fast_tanh(acc[j][r] + bbias) + hidden[idx];
        }
    }
}

// ---------- launch ----------
extern "C" void kernel_launch(void* const* d_in, const int* in_sizes, int n_in,
                              void* d_out, int out_size, void* d_ws, size_t ws_size,
                              hipStream_t stream) {
    const float* target = (const float*)d_in[0];   // (B, T, 2H)
    const float* hidden = (const float*)d_in[1];   // (B, S, 2H)
    const float* W      = (const float*)d_in[2];   // (4H, 2H)
    const float* bias   = (const float*)d_in[3];   // (2H,)
    float* out = (float*)d_out;

    uint8_t* ws = (uint8_t*)d_ws;
    uint32_t* catU  = (uint32_t*)ws;                                   // 128 MB
    uint32_t* tgtU  = (uint32_t*)(ws + (size_t)128 * 1024 * 1024);     //   4 MB
    uint32_t* tgtTU = (uint32_t*)(ws + (size_t)132 * 1024 * 1024);     //   4 MB
    uint32_t* wTU   = (uint32_t*)(ws + (size_t)136 * 1024 * 1024);     //   4 MB

    prep_hidden_kernel<<<4096, 256, 0, stream>>>(hidden, catU);
    prep_tgt_kernel  <<<1024, 256, 0, stream>>>(target, tgtU);
    prep_tgtT_kernel <<<1024, 256, 0, stream>>>(target, tgtTU);
    prep_wT_kernel   <<<1024, 256, 0, stream>>>(W, wTU);

    attn_kernel<<<512, 128, 0, stream>>>(catU, tgtU, tgtTU);
    gemm_kernel<<<dim3(512, 16), 128, 0, stream>>>(catU, wTU, bias, hidden, out);
}
